// MambaBlock_68882685493649
// MI455X (gfx1250) — matrix operand order repaired and checked by reading, not compiled, not run on hardware
//
#include <hip/hip_runtime.h>
#include <hip/hip_bf16.h>
#include <math.h>

// ---------------- problem constants ----------------
#define B_    4
#define L_    2048
#define DM    1024        // d_model
#define DI    2048        // d_inner
#define DS    16          // d_state
#define DC    4           // d_conv
#define DTR   64          // dt_rank
#define MROWS (B_*L_)     // 8192

typedef __bf16 bf16_t;
typedef __attribute__((ext_vector_type(16))) __bf16 v16bf;
typedef __attribute__((ext_vector_type(8)))  __bf16 v8bf;
typedef __attribute__((ext_vector_type(8)))  float  v8f;
typedef __attribute__((ext_vector_type(4)))  unsigned int u32x4;
typedef __attribute__((ext_vector_type(8)))  int  i32x8;
typedef __attribute__((ext_vector_type(4)))  int  i32x4;

__device__ __forceinline__ float silu_f(float x) { return x / (1.0f + __expf(-x)); }

// ---------------- WMMA fragment helpers (contiguous k order, wave32) ----------------
// A: 16x32 bf16, row-major in memory (lda = K).
// lane&15 = M row; lanes 0-15 hold K = kb+{0..15}; lanes 16-31 hold K = kb+{16..31}.
__device__ __forceinline__ v16bf load_a_frag(const bf16_t* __restrict__ A, int lda,
                                             int mbase, int kb, int lane) {
    const int m  = mbase + (lane & 15);
    const int k0 = kb + ((lane >> 4) << 4);
    const bf16_t* p = A + (size_t)m * lda + k0;
    v8bf lo = *(const v8bf*)(p);        // K = k0 .. k0+7
    v8bf hi = *(const v8bf*)(p + 8);    // K = k0+8 .. k0+15
    v16bf r;
#pragma unroll
    for (int i = 0; i < 8; ++i) { r[i] = lo[i]; r[i + 8] = hi[i]; }
    return r;
}

// B fragment from global: 32x16 (KxN), weights W[N][K] row-major so B[k][n] = W[n][k].
__device__ __forceinline__ v16bf load_b_frag(const bf16_t* __restrict__ W, int ldb,
                                             int nbase, int kb, int lane) {
    const int n  = nbase + (lane & 15);
    const int k0 = kb + ((lane >> 4) << 4);
    return *(const v16bf*)(W + (size_t)n * ldb + k0);
}

// B fragment from an LDS-staged tile: tile row r (local n) holds 32 contiguous K bf16 (64B).
__device__ __forceinline__ v16bf lds_b_frag(const bf16_t* bt, int t, int lane) {
    const int n  = 16 * t + (lane & 15);
    const int k0 = (lane >> 4) << 4;
    return *(const v16bf*)(bt + n * 32 + k0);
}

template<int NT>
__device__ __forceinline__ void wmma_rowtile(const bf16_t* __restrict__ A,
                                             const bf16_t* __restrict__ W, int K,
                                             int mbase, int nbase, int lane, v8f acc[NT]) {
    for (int kb = 0; kb < K; kb += 32) {
        v16bf a = load_a_frag(A, K, mbase, kb, lane);
#pragma unroll
        for (int t = 0; t < NT; ++t) {
            v16bf b = load_b_frag(W, K, nbase + 16 * t, kb, lane);
            acc[t] = __builtin_amdgcn_wmma_f32_16x16x32_bf16(
                false, a, false, b, (short)0, acc[t], false, false);
        }
    }
}

// ---------------- Tensor Data Mover: 2-D tile (rows x tile_k bf16) -> LDS ----------------
// Builds D# group0/group1 per CDNA5 ISA sec 8.3/8.4 and issues TENSOR_LOAD_TO_LDS.
// Wave-level op (EXEC ignored); caller must issue from exactly one wave and pair with
// s_wait_tensorcnt + workgroup barrier before consumers read the LDS tile.
__device__ __forceinline__ void tdm_load_tile_2d(unsigned lds_off, const void* gptr,
                                                 unsigned stride_elems, unsigned tile_k,
                                                 unsigned tile_rows) {
    const unsigned long long ga = (unsigned long long)(size_t)gptr;
    u32x4 g0;
    g0[0] = 1u;                                            // count=1, user desc, gather off
    g0[1] = lds_off;                                       // lds_addr (bytes)
    g0[2] = (unsigned)ga;                                  // global_addr[31:0]
    g0[3] = ((unsigned)(ga >> 32) & 0x01FFFFFFu) | (2u << 30); // global_addr[56:32] | type=2
    i32x8 g1;
    g1[0] = (int)(1u << 16);          // workgroup_mask=0, data_size=1 (2 bytes)
    g1[1] = (int)(tile_k << 16);      // tensor_dim0[15:0] (OOB bound = tile)
    g1[2] = (int)(tile_rows << 16);   // tensor_dim0 hi = 0 | tensor_dim1[15:0]
    g1[3] = (int)(tile_k << 16);      // tensor_dim1 hi = 0 | tile_dim0
    g1[4] = (int)tile_rows;           // tile_dim1 ; tile_dim2 = 0
    g1[5] = (int)stride_elems;        // tensor_dim0_stride[31:0] (row pitch, elems)
    g1[6] = 0;                        // stride hi, tensor_dim1_stride lo
    g1[7] = 0;
    i32x4 z4 = {0, 0, 0, 0};
#if __clang_major__ >= 23
    i32x8 z8 = {0, 0, 0, 0, 0, 0, 0, 0};
    __builtin_amdgcn_tensor_load_to_lds(g0, g1, z4, z4, z8, 0);
#else
    __builtin_amdgcn_tensor_load_to_lds(g0, g1, z4, z4, 0);
#endif
}

__device__ __forceinline__ unsigned lds_offset_of(const void* p) {
    return (unsigned)(size_t)p;       // flat LDS aperture: addr[31:0] = LDS byte offset
}

// ---------------- layernorm (fp32) -> bf16 ----------------
__global__ __launch_bounds__(256) void mb_layernorm_kernel(
    const float* __restrict__ x, const float* __restrict__ g,
    const float* __restrict__ b, bf16_t* __restrict__ xnbf) {
    const int row = blockIdx.x;
    const float* xr = x + (size_t)row * DM;
    __shared__ float sred[256];
    float s = 0.f;
    for (int i = threadIdx.x; i < DM; i += 256) s += xr[i];
    sred[threadIdx.x] = s; __syncthreads();
    for (int st = 128; st > 0; st >>= 1) {
        if (threadIdx.x < st) sred[threadIdx.x] += sred[threadIdx.x + st];
        __syncthreads();
    }
    const float mu = sred[0] * (1.0f / DM);
    __syncthreads();
    float v = 0.f;
    for (int i = threadIdx.x; i < DM; i += 256) { float d = xr[i] - mu; v += d * d; }
    sred[threadIdx.x] = v; __syncthreads();
    for (int st = 128; st > 0; st >>= 1) {
        if (threadIdx.x < st) sred[threadIdx.x] += sred[threadIdx.x + st];
        __syncthreads();
    }
    const float rs = rsqrtf(sred[0] * (1.0f / DM) + 1e-5f);
    for (int i = threadIdx.x; i < DM; i += 256)
        xnbf[(size_t)row * DM + i] = (bf16_t)((xr[i] - mu) * rs * g[i] + b[i]);
}

// ---------------- fp32 -> bf16 convert (weights) ----------------
__global__ void mb_f2bf_kernel(const float* __restrict__ s, bf16_t* __restrict__ d, int n) {
    int i = blockIdx.x * blockDim.x + threadIdx.x;
    if (i < n) d[i] = (bf16_t)s[i];
}

// ---------------- GEMM1: xz = xn @ in_proj^T ; split u / silu(z) (TDM-staged B) ----------------
__global__ __launch_bounds__(256) void mb_gemm_inproj_kernel(
    const bf16_t* __restrict__ A, const bf16_t* __restrict__ W,
    float* __restrict__ u, float* __restrict__ zs) {
    __shared__ bf16_t btile[2][64 * 32];      // 2 x 4KB double buffer
    const int lane = threadIdx.x & 31, wave = threadIdx.x >> 5;
    const int mbase = (blockIdx.x * 8 + wave) * 16;
    const int nbase = blockIdx.y * 64;
    const int KSTEPS = DM / 32;
    v8f acc[4];
#pragma unroll
    for (int t = 0; t < 4; ++t) acc[t] = (v8f){0,0,0,0,0,0,0,0};
    if (threadIdx.x < 32)
        tdm_load_tile_2d(lds_offset_of(&btile[0][0]), W + (size_t)nbase * DM, DM, 32, 64);
    for (int i = 0; i < KSTEPS; ++i) {
        const int kb = i * 32;
        if (threadIdx.x < 32) {
            if (i + 1 < KSTEPS) {
                tdm_load_tile_2d(lds_offset_of(&btile[(i + 1) & 1][0]),
                                 W + (size_t)nbase * DM + (kb + 32), DM, 32, 64);
                __builtin_amdgcn_s_wait_tensorcnt(1);   // tile i complete (in-order)
            } else {
                __builtin_amdgcn_s_wait_tensorcnt(0);
            }
        }
        __syncthreads();
        v16bf a = load_a_frag(A, DM, mbase, kb, lane);
        const bf16_t* bt = &btile[i & 1][0];
#pragma unroll
        for (int t = 0; t < 4; ++t) {
            v16bf b = lds_b_frag(bt, t, lane);
            acc[t] = __builtin_amdgcn_wmma_f32_16x16x32_bf16(
                false, a, false, b, (short)0, acc[t], false, false);
        }
        __syncthreads();
    }
    const int mr = mbase + ((lane >> 4) << 3);
    const int nc = nbase + (lane & 15);
#pragma unroll
    for (int t = 0; t < 4; ++t)
#pragma unroll
        for (int r = 0; r < 8; ++r) {
            const int m = mr + r, n = nc + 16 * t;
            const float v = acc[t][r];
            if (n < DI) u[(size_t)m * DI + n] = v;
            else        zs[(size_t)m * DI + (n - DI)] = silu_f(v);
        }
}

// ---------------- depthwise causal conv + bias + silu ----------------
__global__ __launch_bounds__(256) void mb_conv_silu_kernel(
    const float* __restrict__ u, const float* __restrict__ cw, const float* __restrict__ cb,
    float* __restrict__ ucf, bf16_t* __restrict__ ucbf) {
    const size_t idx = (size_t)blockIdx.x * 256 + threadIdx.x;
    if (idx >= (size_t)MROWS * DI) return;
    const int d = (int)(idx % DI);
    const size_t m = idx / DI;          // b*L + l
    const int l = (int)(m % L_);
    float acc = cb[d];
#pragma unroll
    for (int k = 0; k < DC; ++k) {
        const int sh = (DC - 1) - k;
        if (l - sh >= 0) acc += u[(m - (size_t)sh) * DI + d] * cw[d * DC + k];
    }
    const float v = silu_f(acc);
    ucf[idx] = v;
    ucbf[idx] = (bf16_t)v;
}

// ---------------- GEMM2: proj = uc @ x_proj^T  (N = 96, TDM-staged B) ----------------
__global__ __launch_bounds__(256) void mb_gemm_xproj_kernel(
    const bf16_t* __restrict__ A, const bf16_t* __restrict__ W,
    float* __restrict__ proj, bf16_t* __restrict__ dtin) {
    __shared__ bf16_t btile[2][96 * 32];      // 2 x 6KB
    const int lane = threadIdx.x & 31, wave = threadIdx.x >> 5;
    const int mbase = (blockIdx.x * 8 + wave) * 16;
    const int KSTEPS = DI / 32;
    v8f acc[6];
#pragma unroll
    for (int t = 0; t < 6; ++t) acc[t] = (v8f){0,0,0,0,0,0,0,0};
    if (threadIdx.x < 32)
        tdm_load_tile_2d(lds_offset_of(&btile[0][0]), W, DI, 32, 96);
    for (int i = 0; i < KSTEPS; ++i) {
        const int kb = i * 32;
        if (threadIdx.x < 32) {
            if (i + 1 < KSTEPS) {
                tdm_load_tile_2d(lds_offset_of(&btile[(i + 1) & 1][0]),
                                 W + (kb + 32), DI, 32, 96);
                __builtin_amdgcn_s_wait_tensorcnt(1);
            } else {
                __builtin_amdgcn_s_wait_tensorcnt(0);
            }
        }
        __syncthreads();
        v16bf a = load_a_frag(A, DI, mbase, kb, lane);
        const bf16_t* bt = &btile[i & 1][0];
#pragma unroll
        for (int t = 0; t < 6; ++t) {
            v16bf b = lds_b_frag(bt, t, lane);
            acc[t] = __builtin_amdgcn_wmma_f32_16x16x32_bf16(
                false, a, false, b, (short)0, acc[t], false, false);
        }
        __syncthreads();
    }
    const int mr = mbase + ((lane >> 4) << 3);
    const int nc = lane & 15;
#pragma unroll
    for (int t = 0; t < 6; ++t)
#pragma unroll
        for (int r = 0; r < 8; ++r) {
            const int m = mr + r, n = nc + 16 * t;
            const float v = acc[t][r];
            proj[(size_t)m * 96 + n] = v;
            if (n < DTR) dtin[(size_t)m * DTR + n] = (bf16_t)v;
        }
}

// ---------------- GEMM4: dt = softplus(dtin @ dt_proj^T + b)  (K=64, direct) ----------------
__global__ __launch_bounds__(256) void mb_gemm_dtproj_kernel(
    const bf16_t* __restrict__ A, const bf16_t* __restrict__ W,
    const float* __restrict__ bias, float* __restrict__ dt) {
    const int lane = threadIdx.x & 31, wave = threadIdx.x >> 5;
    const int mbase = (blockIdx.x * 8 + wave) * 16;
    const int nbase = blockIdx.y * 64;
    v8f acc[4];
#pragma unroll
    for (int t = 0; t < 4; ++t) acc[t] = (v8f){0,0,0,0,0,0,0,0};
    wmma_rowtile<4>(A, W, DTR, mbase, nbase, lane, acc);
    const int mr = mbase + ((lane >> 4) << 3);
    const int nc = nbase + (lane & 15);
#pragma unroll
    for (int t = 0; t < 4; ++t)
#pragma unroll
        for (int r = 0; r < 8; ++r) {
            const int m = mr + r, n = nc + 16 * t;
            const float s = acc[t][r] + bias[n];
            dt[(size_t)m * DI + n] = (s > 20.0f) ? s : log1pf(expf(s));
        }
}

// ---------------- selective scan: one channel per lane, 16 states in regs ----------------
#define TCH 64
__global__ __launch_bounds__(256) void mb_scan_kernel(
    const float* __restrict__ dt, const float* __restrict__ uc,
    const float* __restrict__ proj, const float* __restrict__ zs,
    const float* __restrict__ A_log, const float* __restrict__ D_param,
    bf16_t* __restrict__ ybf) {
    const int b = blockIdx.x >> 3;                       // 4 batches x 8 channel-chunks
    const int d = ((blockIdx.x & 7) << 8) + threadIdx.x; // channel
    __shared__ float bc[TCH][32];                        // [t][0:16)=B, [16:32)=C
    float Ar[DS], h[DS];
#pragma unroll
    for (int n = 0; n < DS; ++n) { Ar[n] = -__expf(A_log[d * DS + n]); h[n] = 0.f; }
    const float Dp = D_param[d];
    for (int t0 = 0; t0 < L_; t0 += TCH) {
        __syncthreads();
        for (int e = threadIdx.x; e < TCH * 32; e += 256) {
            const int tl = e >> 5, j = e & 31;
            bc[tl][j] = proj[(size_t)(b * L_ + t0 + tl) * 96 + 64 + j];
        }
        __syncthreads();
        for (int tl = 0; tl < TCH; ++tl) {
            const size_t gi = (size_t)(b * L_ + t0 + tl) * DI + d;
            const float dtv = dt[gi];
            const float uv  = uc[gi];
            const float du  = dtv * uv;
            float y = 0.f;
#pragma unroll
            for (int n = 0; n < DS; ++n) {
                h[n] = h[n] * __expf(dtv * Ar[n]) + du * bc[tl][n];
                y += h[n] * bc[tl][16 + n];
            }
            y = (y + uv * Dp) * zs[gi];
            ybf[gi] = (bf16_t)y;
        }
    }
}

// ---------------- GEMM3: out = gelu(y @ out_proj^T) + skip (TDM-staged B) ----------------
__global__ __launch_bounds__(256) void mb_gemm_outproj_kernel(
    const bf16_t* __restrict__ A, const bf16_t* __restrict__ W,
    const float* __restrict__ skip, float* __restrict__ out) {
    __shared__ bf16_t btile[2][64 * 32];
    const int lane = threadIdx.x & 31, wave = threadIdx.x >> 5;
    const int mbase = (blockIdx.x * 8 + wave) * 16;
    const int nbase = blockIdx.y * 64;
    const int KSTEPS = DI / 32;
    v8f acc[4];
#pragma unroll
    for (int t = 0; t < 4; ++t) acc[t] = (v8f){0,0,0,0,0,0,0,0};
    if (threadIdx.x < 32)
        tdm_load_tile_2d(lds_offset_of(&btile[0][0]), W + (size_t)nbase * DI, DI, 32, 64);
    for (int i = 0; i < KSTEPS; ++i) {
        const int kb = i * 32;
        if (threadIdx.x < 32) {
            if (i + 1 < KSTEPS) {
                tdm_load_tile_2d(lds_offset_of(&btile[(i + 1) & 1][0]),
                                 W + (size_t)nbase * DI + (kb + 32), DI, 32, 64);
                __builtin_amdgcn_s_wait_tensorcnt(1);
            } else {
                __builtin_amdgcn_s_wait_tensorcnt(0);
            }
        }
        __syncthreads();
        v16bf a = load_a_frag(A, DI, mbase, kb, lane);
        const bf16_t* bt = &btile[i & 1][0];
#pragma unroll
        for (int t = 0; t < 4; ++t) {
            v16bf b = lds_b_frag(bt, t, lane);
            acc[t] = __builtin_amdgcn_wmma_f32_16x16x32_bf16(
                false, a, false, b, (short)0, acc[t], false, false);
        }
        __syncthreads();
    }
    const int mr = mbase + ((lane >> 4) << 3);
    const int nc = nbase + (lane & 15);
#pragma unroll
    for (int t = 0; t < 4; ++t)
#pragma unroll
        for (int r = 0; r < 8; ++r) {
            const int m = mr + r, n = nc + 16 * t;
            const float v = acc[t][r];
            const float g = 0.5f * v * (1.0f + erff(v * 0.70710678118654752440f));
            out[(size_t)m * DM + n] = g + skip[(size_t)m * DM + n];
        }
}

// ---------------- host launcher ----------------
extern "C" void kernel_launch(void* const* d_in, const int* in_sizes, int n_in,
                              void* d_out, int out_size, void* d_ws, size_t ws_size,
                              hipStream_t stream) {
    const float* x       = (const float*)d_in[0];
    const float* ngamma  = (const float*)d_in[1];
    const float* nbeta   = (const float*)d_in[2];
    const float* w_in    = (const float*)d_in[3];
    const float* conv_w  = (const float*)d_in[4];
    const float* conv_b  = (const float*)d_in[5];
    const float* w_xp    = (const float*)d_in[6];
    const float* w_dt    = (const float*)d_in[7];
    const float* dt_b    = (const float*)d_in[8];
    const float* A_log   = (const float*)d_in[9];
    const float* D_param = (const float*)d_in[10];
    const float* w_out   = (const float*)d_in[11];
    float* out = (float*)d_out;

    char* ws = (char*)d_ws;
    size_t off = 0;
    auto take = [&](size_t bytes) -> char* {
        char* p = ws + off;
        off += (bytes + 255) & ~(size_t)255;
        return p;
    };
    bf16_t* xnbf = (bf16_t*)take((size_t)MROWS * DM * sizeof(bf16_t));
    bf16_t* w1bf = (bf16_t*)take((size_t)2 * DI * DM * sizeof(bf16_t));
    bf16_t* w2bf = (bf16_t*)take((size_t)96 * DI * sizeof(bf16_t));
    bf16_t* w3bf = (bf16_t*)take((size_t)DI * DTR * sizeof(bf16_t));
    bf16_t* w4bf = (bf16_t*)take((size_t)DM * DI * sizeof(bf16_t));
    float*  u    = (float*) take((size_t)MROWS * DI * sizeof(float));
    float*  zsil = (float*) take((size_t)MROWS * DI * sizeof(float));
    float*  ucf  = (float*) take((size_t)MROWS * DI * sizeof(float));
    bf16_t* ucbf = (bf16_t*)take((size_t)MROWS * DI * sizeof(bf16_t));
    float*  proj = (float*) take((size_t)MROWS * 96 * sizeof(float));
    bf16_t* dtin = (bf16_t*)take((size_t)MROWS * DTR * sizeof(bf16_t));
    float*  dtf  = (float*) take((size_t)MROWS * DI * sizeof(float));
    bf16_t* ybf  = (bf16_t*)take((size_t)MROWS * DI * sizeof(bf16_t));
    (void)ws_size; (void)in_sizes; (void)n_in; (void)out_size;

    // 1. layernorm -> bf16 activations
    mb_layernorm_kernel<<<MROWS, 256, 0, stream>>>(x, ngamma, nbeta, xnbf);
    // 2. weight conversions
    mb_f2bf_kernel<<<(2 * DI * DM + 255) / 256, 256, 0, stream>>>(w_in,  w1bf, 2 * DI * DM);
    mb_f2bf_kernel<<<(96 * DI + 255) / 256,     256, 0, stream>>>(w_xp,  w2bf, 96 * DI);
    mb_f2bf_kernel<<<(DI * DTR + 255) / 256,    256, 0, stream>>>(w_dt,  w3bf, DI * DTR);
    mb_f2bf_kernel<<<(DM * DI + 255) / 256,     256, 0, stream>>>(w_out, w4bf, DM * DI);
    // 3. in_proj GEMM (M=8192, N=4096, K=1024), TDM-staged B -> u, silu(z)
    mb_gemm_inproj_kernel<<<dim3(MROWS / 128, (2 * DI) / 64), 256, 0, stream>>>(xnbf, w1bf, u, zsil);
    // 4. depthwise causal conv + silu
    mb_conv_silu_kernel<<<(int)(((size_t)MROWS * DI) / 256), 256, 0, stream>>>(u, conv_w, conv_b, ucf, ucbf);
    // 5. x_proj GEMM (N=96), TDM-staged B -> proj (dt|B|C), bf16 dt-rank slice
    mb_gemm_xproj_kernel<<<dim3(MROWS / 128, 1), 256, 0, stream>>>(ucbf, w2bf, proj, dtin);
    // 6. dt_proj GEMM (N=2048, K=64) + softplus
    mb_gemm_dtproj_kernel<<<dim3(MROWS / 128, DI / 64), 256, 0, stream>>>(dtin, w3bf, dt_b, dtf);
    // 7. sequential selective scan + D skip + silu(z) gate -> y (bf16)
    mb_scan_kernel<<<B_ * (DI / 256), 256, 0, stream>>>(dtf, ucf, proj, zsil, A_log, D_param, ybf);
    // 8. out_proj GEMM + exact GELU + residual
    mb_gemm_outproj_kernel<<<dim3(MROWS / 128, DM / 64), 256, 0, stream>>>(ybf, w4bf, x, out);
}
